// LSTMDecoder_27384711480133
// MI455X (gfx1250) — matrix operand order repaired and checked by reading, not compiled, not run on hardware
//
#include <hip/hip_runtime.h>

// ---------------------------------------------------------------------------
// Problem constants
// ---------------------------------------------------------------------------
#define B_    128
#define E_    512
#define D_    512
#define V_    275
#define P_    576          // HW*HW = 24*24
#define T_    300
#define TM1_  299
#define VPAD_ 288          // V padded to multiple of 16
#define ZK_   1536         // D + E + D   (xin | h) for fused LSTM GEMM
#define G4_   2048         // 4*D

#define DEV static __device__ __forceinline__

// ---------------------------------------------------------------------------
// WMMA types / helpers (CDNA5: v_wmma_f32_16x16x32_bf16, wave32)
// ---------------------------------------------------------------------------
typedef __attribute__((ext_vector_type(16))) __bf16        v16bf;
typedef __attribute__((ext_vector_type(8)))  float         v8f;
typedef __attribute__((ext_vector_type(8)))  unsigned int  v8u;

DEV unsigned short f2bf(float f) {
    union { float f; unsigned u; } v; v.f = f;
    unsigned r = v.u + 0x7FFFu + ((v.u >> 16) & 1u);   // RNE
    return (unsigned short)(r >> 16);
}
DEV float bf2f(unsigned s) {
    union { unsigned u; float f; } v; v.u = s << 16;
    return v.f;
}
DEV float sigf(float x) { return 1.0f / (1.0f + __expf(-x)); }

// A-fragment: 16x32 bf16 row-major (lda even). Each lane's half-fragment is
// 16 contiguous bytes -> compiler merges to 2x global_load_b128.
//   m = lane%16; VGPR j holds pair (k,k+1):
//   k = (j<4) ? (lane/16)*8 + 2j : 16 + (lane/16)*8 + 2(j-4)
DEV v16bf load_a_frag(const unsigned short* __restrict__ A, int lda, int lane) {
    int m = lane & 15, hf = lane >> 4;
    v8u u;
#pragma unroll
    for (int j = 0; j < 8; ++j) {
        int k = (j < 4) ? (hf * 8 + 2 * j) : (16 + hf * 8 + 2 * (j - 4));
        u[j] = *(const unsigned int*)(A + m * lda + k);
    }
    return __builtin_bit_cast(v16bf, u);
}

// B-fragment from PRE-SWIZZLED weights: tile (32K x 16N) stored as 32 lanes x
// 8 packed dwords, contiguous -> one aligned 32B load (2x global_load_b128).
DEV v16bf load_b_pk(const unsigned int* __restrict__ Bpk, int tileIdx, int lane) {
    v8u u = *(const v8u*)(Bpk + (size_t)tileIdx * 256 + lane * 8);
    return __builtin_bit_cast(v16bf, u);
}

DEV v8f wmma_bf16(v16bf a, v16bf b, v8f c) {
    return __builtin_amdgcn_wmma_f32_16x16x32_bf16(false, a, false, b,
                                                   (short)0, c, false, false);
}

// Packed-B coordinate helper: idx enumerates packed dword pairs.
// tile = idx/256 ; within tile: lane = (idx%256)/8, j = idx%8
// (k, n) of the pair's low element; high element is (k+1, n).
DEV void pk_coords(int idx, int Ntiles, int& k, int& n) {
    int t = idx >> 8;
    int r = idx & 255;
    int lane = r >> 3, j = r & 7;
    int ktile = t / Ntiles, ntile = t % Ntiles;
    n = ntile * 16 + (lane & 15);
    k = ktile * 32 + ((j < 4) ? ((lane >> 4) * 8 + 2 * j) : (16 + (lane >> 4) * 8 + 2 * (j - 4)));
}

// ---------------------------------------------------------------------------
// One-time setup kernels
// ---------------------------------------------------------------------------

// flat_bf[b][p][e] = bf16(feats[b][e][p])
__global__ void kflat(const float* __restrict__ feats, unsigned short* __restrict__ flatbf) {
    size_t idx = (size_t)blockIdx.x * 256 + threadIdx.x;     // < B*P*E
    int e  = (int)(idx % E_);
    size_t bp = idx / E_;
    int b = (int)(bp / P_), p = (int)(bp % P_);
    flatbf[idx] = f2bf(feats[((size_t)b * E_ + e) * P_ + p]);
}

// Packed Ufb (512 x 1024): n<512 -> U_w[k][n], else f_beta_w[k][n-512]
__global__ void kpk_ufb(const float* __restrict__ Uw, const float* __restrict__ fbw,
                        unsigned int* __restrict__ out) {
    int idx = blockIdx.x * 256 + threadIdx.x;                // < 512*1024/2
    int k, n; pk_coords(idx, 1024 / 16, k, n);
    float lo = (n < 512) ? Uw[k * 512 + n]       : fbw[k * 512 + (n - 512)];
    float hi = (n < 512) ? Uw[(k + 1) * 512 + n] : fbw[(k + 1) * 512 + (n - 512)];
    out[idx] = (unsigned)f2bf(lo) | ((unsigned)f2bf(hi) << 16);
}

// Packed Wcomb (1536 x 2048): k<1024 -> w_ih[n][k], else w_hh[n][k-1024]
__global__ void kpk_wcomb(const float* __restrict__ wih, const float* __restrict__ whh,
                          unsigned int* __restrict__ out) {
    int idx = blockIdx.x * 256 + threadIdx.x;                // < 1536*2048/2
    int k, n; pk_coords(idx, G4_ / 16, k, n);
    float lo = (k < 1024)     ? wih[(size_t)n * 1024 + k]
                              : whh[(size_t)n * 512 + (k - 1024)];
    float hi = ((k + 1) < 1024) ? wih[(size_t)n * 1024 + (k + 1)]
                                : whh[(size_t)n * 512 + (k + 1 - 1024)];
    out[idx] = (unsigned)f2bf(lo) | ((unsigned)f2bf(hi) << 16);
}

// Packed fc (512 x 288): n<275 -> fc_w[k][n], else 0
__global__ void kpk_fc(const float* __restrict__ fcw, unsigned int* __restrict__ out) {
    int idx = blockIdx.x * 256 + threadIdx.x;                // < 512*288/2
    int k, n; pk_coords(idx, VPAD_ / 16, k, n);
    float lo = (n < V_) ? fcw[k * V_ + n]       : 0.0f;
    float hi = (n < V_) ? fcw[(k + 1) * V_ + n] : 0.0f;
    out[idx] = (unsigned)f2bf(lo) | ((unsigned)f2bf(hi) << 16);
}

// Packed Ww (512 x 512)
__global__ void kpk_ww(const float* __restrict__ Ww, unsigned int* __restrict__ out) {
    int idx = blockIdx.x * 256 + threadIdx.x;                // < 512*512/2
    int k, n; pk_coords(idx, 512 / 16, k, n);
    out[idx] = (unsigned)f2bf(Ww[k * 512 + n]) |
               ((unsigned)f2bf(Ww[(k + 1) * 512 + n]) << 16);
}

// mean[b][e] = (1/P) * sum_p feats[b][e][p]   (contiguous in p)
__global__ void kmean(const float* __restrict__ feats, float* __restrict__ mean) {
    int idx = blockIdx.x * 256 + threadIdx.x;                // < B*E
    const float* row = feats + (size_t)idx * P_;
    float s = 0.f;
    for (int p = 0; p < P_; ++p) s += row[p];
    mean[idx] = s * (1.0f / P_);
}

// h0/c0 init + h_bf
__global__ void kinit(const float* __restrict__ mean,
                      const float* __restrict__ ihw, const float* __restrict__ ihb,
                      const float* __restrict__ icw, const float* __restrict__ icb,
                      float* __restrict__ h, float* __restrict__ c,
                      unsigned short* __restrict__ hbf) {
    int idx = blockIdx.x * 256 + threadIdx.x;                // < B*D
    int b = idx >> 9, n = idx & 511;
    float hh = ihb[n], cc = icb[n];
    const float* m = mean + b * E_;
    for (int e = 0; e < E_; ++e) {
        float mv = m[e];
        hh += mv * ihw[(size_t)e * D_ + n];
        cc += mv * icw[(size_t)e * D_ + n];
    }
    h[idx] = hh; c[idx] = cc; hbf[idx] = f2bf(hh);
}

// Ws_bf = bf16(flat_bf @ Ww + W_b)  — [73728 x 512] x [512 x 512], WMMA
__global__ void kws_gemm(const unsigned short* __restrict__ A,
                         const unsigned int* __restrict__ Bpk,
                         const float* __restrict__ bias,
                         unsigned short* __restrict__ Cout) {
    int lane = threadIdx.x & 31;
    int w = blockIdx.x * 8 + (threadIdx.x >> 5);             // 36864 waves
    int mtile = w >> 3, n64 = w & 7;
    const unsigned short* A0 = A + (size_t)mtile * 16 * D_;
    v8f acc[4] = {};
    for (int k0 = 0; k0 < 512; k0 += 32) {
        v16bf a = load_a_frag(A0 + k0, D_, lane);
        int tbase = (k0 >> 5) * 32 + n64 * 4;                // Ntiles = 32
#pragma unroll
        for (int nt = 0; nt < 4; ++nt)
            acc[nt] = wmma_bf16(a, load_b_pk(Bpk, tbase + nt, lane), acc[nt]);
    }
    int hf = lane >> 4, n0 = lane & 15;
#pragma unroll
    for (int nt = 0; nt < 4; ++nt)
#pragma unroll
        for (int i = 0; i < 8; ++i) {
            int row = mtile * 16 + i + hf * 8;
            int col = n64 * 64 + nt * 16 + n0;
            Cout[(size_t)row * 512 + col] = f2bf(acc[nt][i] + bias[col]);
        }
}

// dec_len output (as float)
__global__ void kdeclen(const int* __restrict__ lengths, float* __restrict__ out) {
    int b = threadIdx.x;
    out[b] = (float)(lengths[b] - 1);
}

// ---------------------------------------------------------------------------
// Per-timestep kernels
// ---------------------------------------------------------------------------

// [hU | gate] = h_bf @ Ufb + [U_b | f_beta_b];  gate gets sigmoid. WMMA.
__global__ void k1_hu(const unsigned short* __restrict__ hbf,
                      const unsigned int* __restrict__ Bpk,
                      const float* __restrict__ Ub, const float* __restrict__ fbb,
                      float* __restrict__ hU, float* __restrict__ gate) {
    int lane = threadIdx.x & 31;
    int w = blockIdx.x * 8 + (threadIdx.x >> 5);             // 128 waves
    int mtile = w >> 4, n64 = w & 15;
    const unsigned short* A0 = hbf + (size_t)mtile * 16 * D_;
    v8f acc[4] = {};
    for (int k0 = 0; k0 < 512; k0 += 32) {
        v16bf a = load_a_frag(A0 + k0, D_, lane);
        int tbase = (k0 >> 5) * 64 + n64 * 4;                // Ntiles = 64
#pragma unroll
        for (int nt = 0; nt < 4; ++nt)
            acc[nt] = wmma_bf16(a, load_b_pk(Bpk, tbase + nt, lane), acc[nt]);
    }
    int hf = lane >> 4, n0 = lane & 15;
#pragma unroll
    for (int nt = 0; nt < 4; ++nt)
#pragma unroll
        for (int i = 0; i < 8; ++i) {
            int row = mtile * 16 + i + hf * 8;
            int col = n64 * 64 + nt * 16 + n0;
            float v = acc[nt][i];
            if (col < 512) hU[row * 512 + col] = v + Ub[col];
            else           gate[row * 512 + (col - 512)] = sigf(v + fbb[col - 512]);
        }
}

// e[b,p] = v_b + sum_d v_w[d] * tanh(Ws_bf[b,p,d] + hU[b,d])   (1 wave / (b,p))
__global__ void k2_escore(const unsigned short* __restrict__ Ws,
                          const float* __restrict__ hU,
                          const float* __restrict__ vw, const float* __restrict__ vb,
                          float* __restrict__ eout) {
    int lane = threadIdx.x & 31;
    int w = blockIdx.x * 8 + (threadIdx.x >> 5);             // 73728 waves
    int b = w / P_;
    const unsigned short* wsrow = Ws + (size_t)w * D_ + lane * 16;
    const float* hrow = hU + b * D_ + lane * 16;
    const float* vrow = vw + lane * 16;
    __builtin_prefetch(wsrow + 8 * D_, 0, 1);                // global_prefetch_b8
    float s = 0.f;
#pragma unroll
    for (int i = 0; i < 8; ++i) {
        unsigned w2 = ((const unsigned*)wsrow)[i];
        float x0 = bf2f(w2 & 0xFFFFu) + hrow[2 * i];
        float x1 = bf2f(w2 >> 16)     + hrow[2 * i + 1];
        s += vrow[2 * i] * tanhf(x0) + vrow[2 * i + 1] * tanhf(x1);
    }
    for (int off = 16; off > 0; off >>= 1) s += __shfl_xor(s, off, 32);
    if (lane == 0) eout[w] = s + vb[0];
}

// softmax over P, write masked alphas, ctx = alpha@flat, assemble z = [emb | gate*ctx | h]
__global__ void k3_soft_ctx(const float* __restrict__ e,
                            const unsigned short* __restrict__ flatbf,
                            const float* __restrict__ gate,
                            const unsigned short* __restrict__ hbf,
                            const float* __restrict__ emb,
                            const int* __restrict__ captions,
                            const int* __restrict__ lengths,
                            int tstep,
                            float* __restrict__ alph_out,
                            unsigned short* __restrict__ zbf) {
    __shared__ float sa[P_];
    __shared__ float sr[256];
    int b = blockIdx.x, tid = threadIdx.x;
    float lmax = -1e30f;
    for (int i = tid; i < P_; i += 256) { float v = e[b * P_ + i]; sa[i] = v; lmax = fmaxf(lmax, v); }
    sr[tid] = lmax; __syncthreads();
    for (int s = 128; s > 0; s >>= 1) { if (tid < s) sr[tid] = fmaxf(sr[tid], sr[tid + s]); __syncthreads(); }
    float mx = sr[0]; __syncthreads();
    float lsum = 0.f;
    for (int i = tid; i < P_; i += 256) { float v = __expf(sa[i] - mx); sa[i] = v; lsum += v; }
    sr[tid] = lsum; __syncthreads();
    for (int s = 128; s > 0; s >>= 1) { if (tid < s) sr[tid] += sr[tid + s]; __syncthreads(); }
    float inv = 1.0f / sr[0]; __syncthreads();

    int dec = lengths[b] - 1;
    float m = (dec > tstep) ? 1.0f : 0.0f;
    for (int i = tid; i < P_; i += 256) {
        float a = sa[i] * inv; sa[i] = a;
        alph_out[(size_t)b * TM1_ * P_ + (size_t)tstep * P_ + i] = a * m;
    }
    __syncthreads();

    int tok = captions[b * T_ + tstep];
    for (int ed = tid; ed < E_; ed += 256) {
        float acc = 0.f;
        const unsigned short* col = flatbf + (size_t)b * P_ * E_ + ed;
        for (int p = 0; p < P_; ++p) acc += sa[p] * bf2f(col[(size_t)p * E_]);
        unsigned short* z = zbf + (size_t)b * ZK_;
        z[512 + ed]  = f2bf(gate[b * E_ + ed] * acc);
        z[ed]        = f2bf(emb[(size_t)tok * D_ + ed]);
        z[1024 + ed] = hbf[b * D_ + ed];
    }
}

// g = z_bf @ Wcomb  — [128 x 1536] x [1536 x 2048], WMMA
__global__ void k4_lstm_gemm(const unsigned short* __restrict__ zbf,
                             const unsigned int* __restrict__ Bpk,
                             float* __restrict__ g) {
    int lane = threadIdx.x & 31;
    int w = blockIdx.x * 8 + (threadIdx.x >> 5);             // 256 waves
    int mtile = w >> 5, n64 = w & 31;
    const unsigned short* A0 = zbf + (size_t)mtile * 16 * ZK_;
    v8f acc[4] = {};
    for (int k0 = 0; k0 < ZK_; k0 += 32) {
        v16bf a = load_a_frag(A0 + k0, ZK_, lane);
        int tbase = (k0 >> 5) * 128 + n64 * 4;               // Ntiles = 128
#pragma unroll
        for (int nt = 0; nt < 4; ++nt)
            acc[nt] = wmma_bf16(a, load_b_pk(Bpk, tbase + nt, lane), acc[nt]);
    }
    int hf = lane >> 4, n0 = lane & 15;
#pragma unroll
    for (int nt = 0; nt < 4; ++nt)
#pragma unroll
        for (int i = 0; i < 8; ++i) {
            int row = mtile * 16 + i + hf * 8;
            int col = n64 * 64 + nt * 16 + n0;
            g[(size_t)row * G4_ + col] = acc[nt][i];
        }
}

// LSTM cell pointwise: gate order i, f, g, o
__global__ void k5_cell(const float* __restrict__ g,
                        const float* __restrict__ bih, const float* __restrict__ bhh,
                        float* __restrict__ h, float* __restrict__ c,
                        unsigned short* __restrict__ hbf) {
    int idx = blockIdx.x * 256 + threadIdx.x;                // < B*D
    int b = idx >> 9, d = idx & 511;
    const float* gr = g + (size_t)b * G4_;
    float ig = gr[d]          + bih[d]          + bhh[d];
    float fg = gr[512 + d]    + bih[512 + d]    + bhh[512 + d];
    float gg = gr[1024 + d]   + bih[1024 + d]   + bhh[1024 + d];
    float og = gr[1536 + d]   + bih[1536 + d]   + bhh[1536 + d];
    float c2 = sigf(fg) * c[idx] + sigf(ig) * tanhf(gg);
    float h2 = sigf(og) * tanhf(c2);
    c[idx] = c2; h[idx] = h2; hbf[idx] = f2bf(h2);
}

// preds[:, t, :] = (h2 @ fc + fc_b) * mask   — [128 x 512] x [512 x 288], WMMA
__global__ void k6_pred(const unsigned short* __restrict__ hbf,
                        const unsigned int* __restrict__ Bpk,
                        const float* __restrict__ fc_bias,
                        const int* __restrict__ lengths, int tstep,
                        float* __restrict__ preds) {
    int lane = threadIdx.x & 31;
    int w = blockIdx.x * 8 + (threadIdx.x >> 5);             // 72 waves
    int mtile = w / 9, n32 = w % 9;
    const unsigned short* A0 = hbf + (size_t)mtile * 16 * D_;
    v8f acc[2] = {};
    for (int k0 = 0; k0 < 512; k0 += 32) {
        v16bf a = load_a_frag(A0 + k0, D_, lane);
        int tbase = (k0 >> 5) * 18 + n32 * 2;                // Ntiles = 18
#pragma unroll
        for (int nt = 0; nt < 2; ++nt)
            acc[nt] = wmma_bf16(a, load_b_pk(Bpk, tbase + nt, lane), acc[nt]);
    }
    int hf = lane >> 4, n0 = lane & 15;
#pragma unroll
    for (int nt = 0; nt < 2; ++nt)
#pragma unroll
        for (int i = 0; i < 8; ++i) {
            int row = mtile * 16 + i + hf * 8;
            int col = n32 * 32 + nt * 16 + n0;
            if (col < V_) {
                float m = ((lengths[row] - 1) > tstep) ? 1.0f : 0.0f;
                preds[(size_t)row * TM1_ * V_ + (size_t)tstep * V_ + col] =
                    (acc[nt][i] + fc_bias[col]) * m;
            }
        }
}

// ---------------------------------------------------------------------------
// Host launcher
// ---------------------------------------------------------------------------
extern "C" void kernel_launch(void* const* d_in, const int* in_sizes, int n_in,
                              void* d_out, int out_size, void* d_ws, size_t ws_size,
                              hipStream_t stream) {
    (void)in_sizes; (void)n_in; (void)out_size; (void)ws_size;
    const float* feats    = (const float*)d_in[0];
    const int*   captions = (const int*)d_in[1];
    const int*   lengths  = (const int*)d_in[2];
    const float* U_w      = (const float*)d_in[3];
    const float* U_b      = (const float*)d_in[4];
    const float* W_w      = (const float*)d_in[5];
    const float* W_b      = (const float*)d_in[6];
    const float* v_w      = (const float*)d_in[7];
    const float* v_b      = (const float*)d_in[8];
    const float* ih_w     = (const float*)d_in[9];
    const float* ih_b     = (const float*)d_in[10];
    const float* ic_w     = (const float*)d_in[11];
    const float* ic_b     = (const float*)d_in[12];
    const float* fb_w     = (const float*)d_in[13];
    const float* fb_b     = (const float*)d_in[14];
    const float* fc_w     = (const float*)d_in[15];
    const float* fc_b     = (const float*)d_in[16];
    const float* emb      = (const float*)d_in[17];
    const float* w_ih     = (const float*)d_in[18];
    const float* w_hh     = (const float*)d_in[19];
    const float* b_ih     = (const float*)d_in[20];
    const float* b_hh     = (const float*)d_in[21];

    // workspace carve-out (256B aligned)
    char* cur = (char*)d_ws;
    auto alloc = [&](size_t bytes) -> void* {
        void* p = (void*)cur;
        cur += (bytes + 255) & ~(size_t)255;
        return p;
    };
    unsigned short* flat_bf = (unsigned short*)alloc((size_t)B_ * P_ * E_ * 2);
    unsigned short* Ws_bf   = (unsigned short*)alloc((size_t)B_ * P_ * D_ * 2);
    unsigned int*   Ufb_pk  = (unsigned int*)alloc((size_t)512 * 1024 * 2);
    unsigned int*   Wc_pk   = (unsigned int*)alloc((size_t)ZK_ * G4_ * 2);
    unsigned int*   fc_pk   = (unsigned int*)alloc((size_t)512 * VPAD_ * 2);
    unsigned int*   Ww_pk   = (unsigned int*)alloc((size_t)512 * 512 * 2);
    float* mean  = (float*)alloc((size_t)B_ * E_ * 4);
    float* h     = (float*)alloc((size_t)B_ * D_ * 4);
    float* c     = (float*)alloc((size_t)B_ * D_ * 4);
    unsigned short* h_bf = (unsigned short*)alloc((size_t)B_ * D_ * 2);
    float* hU    = (float*)alloc((size_t)B_ * D_ * 4);
    float* gate  = (float*)alloc((size_t)B_ * E_ * 4);
    float* esc   = (float*)alloc((size_t)B_ * P_ * 4);
    unsigned short* z_bf = (unsigned short*)alloc((size_t)B_ * ZK_ * 2);
    float* g     = (float*)alloc((size_t)B_ * G4_ * 4);

    // output layout: preds | alphas | dec_len
    const size_t PREDS_N  = (size_t)B_ * TM1_ * V_;
    const size_t ALPHAS_N = (size_t)B_ * TM1_ * P_;
    float* preds_out = (float*)d_out;
    float* alph_out  = preds_out + PREDS_N;
    float* dec_out   = preds_out + PREDS_N + ALPHAS_N;

    // --- one-time setup -----------------------------------------------------
    kflat    <<<147456, 256, 0, stream>>>(feats, flat_bf);
    kpk_ufb  <<<1024,   256, 0, stream>>>(U_w, fb_w, Ufb_pk);     // 512*1024/2 pairs
    kpk_wcomb<<<6144,   256, 0, stream>>>(w_ih, w_hh, Wc_pk);     // 1536*2048/2
    kpk_fc   <<<288,    256, 0, stream>>>(fc_w, fc_pk);           // 512*288/2
    kpk_ww   <<<512,    256, 0, stream>>>(W_w, Ww_pk);            // 512*512/2
    kmean    <<<256,    256, 0, stream>>>(feats, mean);
    kinit    <<<256,    256, 0, stream>>>(mean, ih_w, ih_b, ic_w, ic_b, h, c, h_bf);
    kws_gemm <<<4608,   256, 0, stream>>>(flat_bf, Ww_pk, W_b, Ws_bf);
    kdeclen  <<<1,      128, 0, stream>>>(lengths, dec_out);

    // --- sequential decode: 299 steps --------------------------------------
    for (int t = 0; t < TM1_; ++t) {
        k1_hu       <<<16,   256, 0, stream>>>(h_bf, Ufb_pk, U_b, fb_b, hU, gate);
        k2_escore   <<<9216, 256, 0, stream>>>(Ws_bf, hU, v_w, v_b, esc);
        k3_soft_ctx <<<B_,   256, 0, stream>>>(esc, flat_bf, gate, h_bf, emb,
                                               captions, lengths, t, alph_out, z_bf);
        k4_lstm_gemm<<<32,   256, 0, stream>>>(z_bf, Wc_pk, g);
        k5_cell     <<<256,  256, 0, stream>>>(g, b_ih, b_hh, h, c, h_bf);
        k6_pred     <<<9,    256, 0, stream>>>(h_bf, fc_pk, fc_b, lengths, t, preds_out);
    }
}
